// MolConv3_2233382994426
// MI455X (gfx1250) — compile-verified, matrix operand order checked by reading, not run on hardware
//
#include <hip/hip_runtime.h>
#include <hip/hip_bf16.h>
#include <math.h>

typedef __attribute__((ext_vector_type(2)))  float        v2f;
typedef __attribute__((ext_vector_type(8)))  float        v8f;
typedef __attribute__((ext_vector_type(16))) __bf16       v16bf;
typedef __attribute__((ext_vector_type(4)))  unsigned int u32x4;
typedef __attribute__((ext_vector_type(8)))  int          i32x8;
typedef __attribute__((ext_vector_type(4)))  int          i32x4;

enum : int { Bt = 64, Nt = 512, Kt = 32, Dt = 21, C1t = 53, Ot = 64 };

__device__ __forceinline__ float sigmf(float v) { return 1.f / (1.f + __expf(-v)); }

// ---------------------------------------------------------------------------
// K1: pairwise distances (f32 WMMA 16x16x4) + top-K selection per point.
//   grid = B * (N/32), block = 256 (8 waves). LDS: x tile (TDM), xx, pd tile.
// ---------------------------------------------------------------------------
__global__ __launch_bounds__(256) void k1_pd_topk(const float* __restrict__ x,
                                                  int* __restrict__ idx_out,
                                                  float* __restrict__ dist_out) {
  __shared__ float smem[24 * 512 + 512 + 32 * 512];  // lx | lxx | lpd
  float* lx  = smem;                 // [24][512], rows 21..23 zero padding
  float* lxx = smem + 24 * 512;      // [512]
  float* lpd = smem + 24 * 512 + 512;// [32][512]

  const int b    = blockIdx.x >> 4;
  const int n0   = (blockIdx.x & 15) * 32;
  const int tid  = threadIdx.x;
  const int lane = tid & 31, wave = tid >> 5;
  const bool hi  = lane >= 16;
  const int li   = lane & 15;
  const float* xb = x + (size_t)b * Dt * Nt;

#if __has_builtin(__builtin_amdgcn_tensor_load_to_lds)
  // Tensor Data Mover: 2D tile 512x21 f32 from global -> LDS offset 0.
  if (wave == 0) {
    unsigned long long ga = (unsigned long long)(const void*)xb;
    u32x4 g0 = { 1u,                                   // count=1
                 0u,                                   // lds_addr=0
                 (unsigned)ga,
                 (unsigned)((ga >> 32) & 0x01FFFFFFu) | 0x80000000u }; // type=2
    i32x8 g1 = { (int)(2u << 16),            // data_size = 4B
                 (int)(512u << 16),          // tensor_dim0[15:0]
                 (int)(21u  << 16),          // tensor_dim0 hi=0 | tensor_dim1 lo
                 (int)(512u << 16),          // tensor_dim1 hi=0 | tile_dim0
                 21,                         // tile_dim1 | tile_dim2=0
                 512,                        // tensor_dim0_stride lo
                 0, 0 };
    i32x4 gz = {0, 0, 0, 0};
#if __clang_major__ >= 23
    i32x8 gz8 = {0, 0, 0, 0, 0, 0, 0, 0};
    __builtin_amdgcn_tensor_load_to_lds(g0, g1, gz, gz, gz8, 0);
#else
    __builtin_amdgcn_tensor_load_to_lds(g0, g1, gz, gz, 0);
#endif
    __builtin_amdgcn_s_wait_tensorcnt(0);
  }
  for (int i = tid; i < 3 * 512; i += 256) lx[21 * 512 + i] = 0.f; // pad rows
#else
  for (int i = tid; i < 24 * 512; i += 256) lx[i] = (i < 21 * 512) ? xb[i] : 0.f;
#endif
  __syncthreads();

  for (int c = tid; c < 512; c += 256) {
    float s = 0.f;
    for (int d = 0; d < Dt; ++d) { float v = lx[d * 512 + c]; s += v * v; }
    lxx[c] = s;
  }
  __syncthreads();

  // pd tile: 32 rows (n0..n0+31) x 512 cols via WMMA f32 16x16x4.
  const int r = wave & 1;  // row group 0/1
  for (int ii = 0; ii < 8; ++ii) {
    const int ct = (wave >> 1) + 4 * ii;
    const int c0 = ct * 16;
    v8f acc = {0.f, 0.f, 0.f, 0.f, 0.f, 0.f, 0.f, 0.f};
#if __has_builtin(__builtin_amdgcn_wmma_f32_16x16x4_f32)
    const int arow = n0 + 16 * r + li;
#pragma unroll
    for (int kk = 0; kk < 6; ++kk) {
      const int d0 = 4 * kk + (hi ? 2 : 0);
      v2f a, bb;
      a.x  = lx[d0 * 512 + arow];
      a.y  = lx[(d0 + 1) * 512 + arow];
      bb.x = lx[d0 * 512 + c0 + li];
      bb.y = lx[(d0 + 1) * 512 + c0 + li];
      acc = __builtin_amdgcn_wmma_f32_16x16x4_f32(false, a, false, bb,
                                                  (short)0, acc, false, false);
    }
#else
    for (int v = 0; v < 8; ++v) {
      const int rowp = n0 + 16 * r + v + (hi ? 8 : 0);
      float s = 0.f;
      for (int d = 0; d < Dt; ++d) s += lx[d * 512 + rowp] * lx[d * 512 + c0 + li];
      acc[v] = s;
    }
#endif
#pragma unroll
    for (int v = 0; v < 8; ++v) {
      const int rr = 16 * r + v + (hi ? 8 : 0);
      lpd[rr * 512 + c0 + li] = 2.f * acc[v] - lxx[n0 + rr] - lxx[c0 + li];
    }
  }
  __syncthreads();

  // top-K=32 per row: iterative wave argmax, ties -> lowest index (jax top_k).
  for (int j = 0; j < 4; ++j) {
    const int rr = wave * 4 + j;
    const int n  = n0 + rr;
    float* row = lpd + rr * 512;
    for (int k = 0; k < Kt; ++k) {
      float bv = -3.4e38f; int bi = 1 << 30;
      for (int c = lane; c < 512; c += 32) {
        float v = row[c];
        if (v > bv || (v == bv && c < bi)) { bv = v; bi = c; }
      }
#pragma unroll
      for (int off = 16; off > 0; off >>= 1) {
        float ov = __shfl_xor(bv, off, 32);
        int   oi = __shfl_xor(bi, off, 32);
        if (ov > bv || (ov == bv && oi < bi)) { bv = ov; bi = oi; }
      }
      if (lane == (bi & 31)) row[bi] = -3.4e38f;
      if (lane == 0) {
        dist_out[((size_t)b * Nt + n) * Kt + k] = -bv;
        idx_out [((size_t)b * Nt + n) * Kt + k] = bi;
      }
    }
  }
}

// ---------------------------------------------------------------------------
// K2: G[...,k,0] = dot(gf_k, gf_0); accumulate S1[b,k] = sum_n G^2.
// ---------------------------------------------------------------------------
__global__ __launch_bounds__(256) void k2_gcol(const float* __restrict__ x,
                                               const int* __restrict__ idx,
                                               float* __restrict__ gcol0,
                                               float* __restrict__ S1) {
  __shared__ float s1p[32];
  const int b = blockIdx.x >> 4;
  const int nb = (blockIdx.x & 15) * 32;
  const int tid = threadIdx.x, lane = tid & 31, wave = tid >> 5;
  if (tid < 32) s1p[tid] = 0.f;
  __syncthreads();
  const float* xb = x + (size_t)b * Dt * Nt;
  float part = 0.f;
  for (int j = 0; j < 4; ++j) {
    const int n = nb + wave * 4 + j;
    const size_t base = ((size_t)b * Nt + n) * Kt;
    const int ig = idx[base + lane];
    float dot = 0.f;
#pragma unroll
    for (int d = 0; d < Dt; ++d) {
      float v = xb[d * Nt + ig];
      dot += v * __shfl(v, 0, 32);   // lane 0 holds neighbor-0 features
    }
    gcol0[base + lane] = dot;
    part += dot * dot;
  }
  atomicAdd(&s1p[lane], part);
  __syncthreads();
  if (tid < 32) atomicAdd(&S1[b * Kt + tid], s1p[tid]);
}

// ---------------------------------------------------------------------------
// K3: per-b: sub, invr2 table (32x32), dist LN stats, counts, hc + LN stats.
// ---------------------------------------------------------------------------
__global__ __launch_bounds__(256) void k3_perb(
    const float* __restrict__ x, const unsigned char* __restrict__ mask,
    const float* __restrict__ w_center, const float* __restrict__ gcol0,
    const float* __restrict__ S1, const float* __restrict__ dist,
    float* __restrict__ sub, float* __restrict__ invr2, float* __restrict__ hc,
    float* __restrict__ hcst, float* __restrict__ dst, float* __restrict__ cnt) {
  __shared__ float inv1[32];
  __shared__ float s2a[1024];
  __shared__ float red[256];
  const int b = blockIdx.x, tid = threadIdx.x, lane = tid & 31, wave = tid >> 5;
  if (tid < 32) inv1[tid] = 1.f / fmaxf(sqrtf(S1[b * Kt + tid]), 1e-12f);
  for (int i = tid; i < 1024; i += 256) s2a[i] = 0.f;
  __syncthreads();

  const size_t nkb = (size_t)b * Nt * Kt;
  for (int i = tid; i < Nt * Kt; i += 256)
    sub[nkb + i] = gcol0[nkb + i] * inv1[i & 31];

  float racc[32];
#pragma unroll
  for (int j = 0; j < 32; ++j) racc[j] = 0.f;
  for (int n = wave; n < Nt; n += 8) {
    float g = gcol0[nkb + (size_t)n * Kt + lane] * inv1[lane];
#pragma unroll
    for (int j = 0; j < 32; ++j) {
      float p = g * __shfl(g, j, 32);
      racc[j] += p * p;
    }
  }
#pragma unroll
  for (int j = 0; j < 32; ++j) atomicAdd(&s2a[lane * 32 + j], racc[j]);
  __syncthreads();
  for (int i = tid; i < 1024; i += 256)
    invr2[b * 1024 + i] = 1.f / fmaxf(sqrtf(s2a[i]), 1e-12f);

  // dist LN stats
  float s = 0.f, q = 0.f;
  for (int i = tid; i < Nt * Kt; i += 256) { float v = dist[nkb + i]; s += v; q += v * v; }
  red[tid] = s; __syncthreads();
  for (int o = 128; o > 0; o >>= 1) { if (tid < o) red[tid] += red[tid + o]; __syncthreads(); }
  if (tid == 0) dst[b * 2] = red[0];
  __syncthreads();
  red[tid] = q; __syncthreads();
  for (int o = 128; o > 0; o >>= 1) { if (tid < o) red[tid] += red[tid + o]; __syncthreads(); }
  if (tid == 0) dst[b * 2 + 1] = red[0];
  __syncthreads();

  // mask counts
  float c = 0.f;
  for (int i = tid; i < Nt; i += 256) c += mask[b * Nt + i] ? 1.f : 0.f;
  red[tid] = c; __syncthreads();
  for (int o = 128; o > 0; o >>= 1) { if (tid < o) red[tid] += red[tid + o]; __syncthreads(); }
  if (tid == 0) cnt[b] = red[0];
  __syncthreads();

  // hc = w_center @ x[b]  (53 x 512) + LN stats
  const float* xb = x + (size_t)b * Dt * Nt;
  float hs = 0.f, hq = 0.f;
  for (int i = tid; i < C1t * Nt; i += 256) {
    const int cc = i >> 9, n = i & 511;
    float h = 0.f;
#pragma unroll
    for (int d = 0; d < Dt; ++d) h += w_center[cc * Dt + d] * xb[d * Nt + n];
    hc[(size_t)b * C1t * Nt + i] = h;
    hs += h; hq += h * h;
  }
  red[tid] = hs; __syncthreads();
  for (int o = 128; o > 0; o >>= 1) { if (tid < o) red[tid] += red[tid + o]; __syncthreads(); }
  if (tid == 0) hcst[b * 2] = red[0];
  __syncthreads();
  red[tid] = hq; __syncthreads();
  for (int o = 128; o > 0; o >>= 1) { if (tid < o) red[tid] += red[tid + o]; __syncthreads(); }
  if (tid == 0) hcst[b * 2 + 1] = red[0];
}

// ---------------------------------------------------------------------------
// K4: h_pre = W_update(64x53) @ feat(53x(N*K)) per b, bf16 WMMA 16x16x32,
//     feat built on the fly; accumulate global LN stats.
//     grid = B*8 blocks, 256 thr; each wave handles 16 column tiles.
// ---------------------------------------------------------------------------
__global__ __launch_bounds__(256) void k4_update(
    const float* __restrict__ x, const int* __restrict__ idx,
    const float* __restrict__ dist, const float* __restrict__ sub,
    const float* __restrict__ invr2, const float* __restrict__ hc,
    const float* __restrict__ hcst, const float* __restrict__ dst,
    const float* __restrict__ w_dist, const float* __restrict__ g_dist,
    const float* __restrict__ b_dist, const float* __restrict__ g_center,
    const float* __restrict__ b_center, const float* __restrict__ w_update,
    float* __restrict__ hpre, float* __restrict__ hst) {
  __shared__ float linv[1024];
  __shared__ float hacc[2];
  const int b = blockIdx.x >> 3, chunk = blockIdx.x & 7;
  const int tid = threadIdx.x, lane = tid & 31, wave = tid >> 5;
  const bool hi = lane >= 16;
  const int li = lane & 15;
  for (int i = tid; i < 1024; i += 256) linv[i] = invr2[b * 1024 + i];
  if (tid < 2) hacc[tid] = 0.f;
  __syncthreads();

  const float wd = w_dist[0];
  const float inv_nk = 1.f / (float)(Nt * Kt);
  const float mu_d = dst[b * 2] * inv_nk;
  const float var_d = fmaxf(dst[b * 2 + 1] * inv_nk - mu_d * mu_d, 0.f);
  const float rs_d = rsqrtf(wd * wd * var_d + 1e-5f);
  const float inv_cn = 1.f / (float)(C1t * Nt);
  const float mu_c = hcst[b * 2] * inv_cn;
  const float var_c = fmaxf(hcst[b * 2 + 1] * inv_cn - mu_c * mu_c, 0.f);
  const float rs_c = rsqrtf(var_c + 1e-5f);

  // A operands: W_update in bf16 WMMA layout (4 row tiles x 2 k-steps).
  v16bf A[4][2];
#pragma unroll
  for (int R = 0; R < 4; ++R)
#pragma unroll
    for (int S = 0; S < 2; ++S)
#pragma unroll
      for (int e = 0; e < 16; ++e) {
        const int v = e >> 1;
        const int c = ((v < 4) ? 2 * v : 8 + 2 * v) + (e & 1) + (hi ? 8 : 0) + S * 32;
        const int o = R * 16 + li;
        A[R][S][e] = (__bf16)((c < C1t) ? w_update[o * C1t + c] : 0.f);
      }

  float ls = 0.f, lq = 0.f;
  for (int it = 0; it < 16; ++it) {
    const int t = chunk * 128 + wave + 8 * it;   // 0..1023 column tiles
    const int n = t >> 1, k0 = (t & 1) * 16;
    const size_t bn = (size_t)b * Nt + n;
    const int kk = k0 + li;
    const int ig = idx[bn * Kt + kk];
    __builtin_prefetch(&x[((size_t)b * Dt) * Nt + ig], 0, 1);
    const float dv = dist[bn * Kt + kk];
    const float wl = sigmf((wd * dv - wd * mu_d) * rs_d * g_dist[n * Kt + kk] +
                           b_dist[n * Kt + kk]);
    const float subk = sub[bn * Kt + kk];
    const float sr = sub[bn * Kt + lane];  // lane covers j=0..31

    v16bf Bm[2];
#pragma unroll
    for (int S = 0; S < 2; ++S)
#pragma unroll
      for (int e = 0; e < 16; ++e) {
        const int v = e >> 1;
        const int c = ((v < 4) ? 2 * v : 8 + 2 * v) + (e & 1) + (hi ? 8 : 0) + S * 32;
        float fv = 0.f;
        if (c < C1t) {
          const float hcn = (hc[(size_t)b * C1t * Nt + c * Nt + n] - mu_c) * rs_c;
          const size_t gi = ((size_t)c * Nt + n) * Kt + kk;
          const float fcv = sigmf(hcn * g_center[gi] + b_center[gi]);
          float fn;
          if (c < Dt) fn = x[((size_t)b * Dt + c) * Nt + ig];
          else {
            const int j = c - Dt;
            fn = subk * __shfl(sr, j, 32) * linv[kk * 32 + j];
          }
          fv = wl * fn + fcv;
        }
        Bm[S][e] = (__bf16)fv;
      }

#pragma unroll
    for (int R = 0; R < 4; ++R) {
      v8f acc = {0.f, 0.f, 0.f, 0.f, 0.f, 0.f, 0.f, 0.f};
      acc = __builtin_amdgcn_wmma_f32_16x16x32_bf16(false, A[R][0], false, Bm[0],
                                                    (short)0, acc, false, false);
      acc = __builtin_amdgcn_wmma_f32_16x16x32_bf16(false, A[R][1], false, Bm[1],
                                                    (short)0, acc, false, false);
#pragma unroll
      for (int v = 0; v < 8; ++v) {
        const int o = R * 16 + v + (hi ? 8 : 0);
        const float val = acc[v];
        hpre[(((size_t)b * Ot + o) * Nt + n) * Kt + k0 + li] = val;
        ls += val; lq += val * val;
      }
    }
  }
  atomicAdd(&hacc[0], ls);
  atomicAdd(&hacc[1], lq);
  __syncthreads();
  if (tid == 0) { atomicAdd(&hst[b * 2], hacc[0]); atomicAdd(&hst[b * 2 + 1], hacc[1]); }
}

// ---------------------------------------------------------------------------
// K5: LN + thresholded softplus + mask + k-reduction -> (B,64,N).
// ---------------------------------------------------------------------------
__global__ __launch_bounds__(256) void k5_out(
    const float* __restrict__ hpre, const float* __restrict__ hst,
    const float* __restrict__ g_update, const float* __restrict__ b_update,
    const unsigned char* __restrict__ mask, const float* __restrict__ cnt,
    float* __restrict__ out) {
  const int i = blockIdx.x * 256 + threadIdx.x;
  const int b = i >> 15, o = (i >> 9) & 63, n = i & 511;
  const float M = (float)Ot * (float)Nt * (float)Kt;
  const float mu = hst[b * 2] / M;
  const float var = fmaxf(hst[b * 2 + 1] / M - mu * mu, 0.f);
  const float rs = rsqrtf(var + 1e-5f);
  const float mk = mask[b * Nt + n] ? 1.f : 0.f;
  const float cc = fmaxf(cnt[b], 0.1f);
  float acc = 0.f;
  const size_t hb = (((size_t)b * Ot + o) * Nt + n) * Kt;
  const size_t gb = ((size_t)o * Nt + n) * Kt;
  for (int k = 0; k < Kt; ++k) {
    const float hn = (hpre[hb + k] - mu) * rs * g_update[gb + k] + b_update[gb + k];
    const float sp = (hn > 20.f) ? hn : log1pf(__expf(hn));
    acc += sp;
  }
  out[i] = mk * acc / cc;
}

// ---------------------------------------------------------------------------
extern "C" void kernel_launch(void* const* d_in, const int* in_sizes, int n_in,
                              void* d_out, int out_size, void* d_ws, size_t ws_size,
                              hipStream_t stream) {
  const float* x            = (const float*)d_in[0];
  // d_in[1] = idx_base (unused by the reference math)
  const unsigned char* mask = (const unsigned char*)d_in[2];
  const float* w_dist   = (const float*)d_in[3];
  const float* g_dist   = (const float*)d_in[4];
  const float* b_dist   = (const float*)d_in[5];
  const float* w_center = (const float*)d_in[6];
  const float* g_center = (const float*)d_in[7];
  const float* b_center = (const float*)d_in[8];
  const float* w_update = (const float*)d_in[9];
  const float* g_update = (const float*)d_in[10];
  const float* b_update = (const float*)d_in[11];
  float* out = (float*)d_out;

  char* ws = (char*)d_ws;
  size_t off = 0;
  auto take = [&](size_t bytes) {
    char* p = ws + off;
    off = (off + bytes + 255) & ~(size_t)255;
    return p;
  };
  const size_t NK = (size_t)Bt * Nt * Kt;
  int*   ws_idx   = (int*)  take(4 * NK);
  float* ws_dist  = (float*)take(4 * NK);
  float* ws_gcol0 = (float*)take(4 * NK);
  float* ws_sub   = (float*)take(4 * NK);
  float* ws_S1    = (float*)take(4ull * Bt * Kt);
  float* ws_invr2 = (float*)take(4ull * Bt * Kt * Kt);
  float* ws_hc    = (float*)take(4ull * Bt * C1t * Nt);
  float* ws_hcst  = (float*)take(4ull * Bt * 2);
  float* ws_dst   = (float*)take(4ull * Bt * 2);
  float* ws_cnt   = (float*)take(4ull * Bt);
  float* ws_hst   = (float*)take(4ull * Bt * 2);
  float* ws_hpre  = (float*)take(4ull * Bt * Ot * Nt * Kt);
  (void)ws_size; (void)in_sizes; (void)n_in; (void)out_size;

  hipMemsetAsync(ws_S1, 0, 4ull * Bt * Kt, stream);
  hipMemsetAsync(ws_hst, 0, 4ull * Bt * 2, stream);

  k1_pd_topk<<<Bt * (Nt / 32), 256, 0, stream>>>(x, ws_idx, ws_dist);
  k2_gcol  <<<Bt * (Nt / 32), 256, 0, stream>>>(x, ws_idx, ws_gcol0, ws_S1);
  k3_perb  <<<Bt, 256, 0, stream>>>(x, mask, w_center, ws_gcol0, ws_S1, ws_dist,
                                    ws_sub, ws_invr2, ws_hc, ws_hcst, ws_dst, ws_cnt);
  k4_update<<<Bt * 8, 256, 0, stream>>>(x, ws_idx, ws_dist, ws_sub, ws_invr2,
                                        ws_hc, ws_hcst, ws_dst, w_dist, g_dist,
                                        b_dist, g_center, b_center, w_update,
                                        ws_hpre, ws_hst);
  k5_out   <<<(Bt * Ot * Nt) / 256, 256, 0, stream>>>(ws_hpre, ws_hst, g_update,
                                                      b_update, mask, ws_cnt, out);
}
